// RNN_seq_gen_26448408609203
// MI455X (gfx1250) — matrix operand order repaired and checked by reading, not compiled, not run on hardware
//
#include <hip/hip_runtime.h>
#include <hip/hip_bf16.h>

typedef __attribute__((ext_vector_type(16))) __bf16 bf16x16;
typedef __attribute__((ext_vector_type(8)))  __bf16 bf16x8;
typedef __attribute__((ext_vector_type(4)))  __bf16 bf16x4;
typedef __attribute__((ext_vector_type(8)))  float  f32x8;
typedef __attribute__((ext_vector_type(4)))  float  f32x4;

#define B_TOT    256
#define DIN      128
#define DLAT     512
#define DOUT     128
#define T_IN     256
#define L_OUT    512
#define NSTEPS   (L_OUT - 1)        // 511 recurrence steps
#define BB       32                 // batch rows per block
#define NBLK     (B_TOT / BB)       // 8 independent blocks (no inter-block sync)
#define NTHREADS 256                // 8 wave32 per block

#define H_STRIDE  528               // 512 + 16 pad (bf16 elems) -> bank-shift rows
#define IN_STRIDE 136               // 128 + 8 pad

// bf16 weight workspace layout (element offsets)
#define WS_WI  0
#define WS_WH  (512 * 128)
#define WS_WD  (WS_WH + 512 * 512)
#define WS_TOT (WS_WD + 128 * 512)

__device__ __forceinline__ f32x8 wmma_bf16(bf16x16 a, bf16x16 b, f32x8 c) {
  // (neg_a, A, neg_b, B, c_mod, C, reuse_a, reuse_b)
  return __builtin_amdgcn_wmma_f32_16x16x32_bf16(false, a, false, b, (short)0, c,
                                                 false, false);
}

// A-fragment (16x32 bf16, rows row0..row0+15) from an LDS row-major buffer.
// ISA layout: lane = row (lane&15); VGPR0-3 hold K = half*8..+7, VGPR4-7 hold
// K = 16 + half*8..+7 (half = lane>>4).
__device__ __forceinline__ bf16x16 load_a_lds(const __bf16* buf, int stride,
                                              int row0, int k0, int lane) {
  const int r  = row0 + (lane & 15);
  const int kh = (lane >> 4) << 3;
  union { bf16x16 f; bf16x8 h[2]; } u;
  u.h[0] = *(const bf16x8*)(buf + r * stride + k0 + kh);
  u.h[1] = *(const bf16x8*)(buf + r * stride + k0 + 16 + kh);
  return u.f;
}

// B-fragment (32x16) for C = A * W^T from row-major W[N][K] in global (bf16).
// ISA layout: lane = column N (lane&15); VGPR0-3 hold K = half*8..+7, VGPR4-7
// hold K = 16 + half*8..+7 (two global_load_b128).
__device__ __forceinline__ bf16x16 load_b_glb(const __bf16* W, int ktot,
                                              int n0, int k0, int lane) {
  const int n  = n0 + (lane & 15);
  const int kh = (lane >> 4) << 3;
  union { bf16x16 f; bf16x8 h[2]; } u;
  u.h[0] = *(const bf16x8*)(W + n * ktot + k0 + kh);
  u.h[1] = *(const bf16x8*)(W + n * ktot + k0 + 16 + kh);
  return u.f;
}

__global__ void cvt_weights_kernel(const float* __restrict__ Wi,
                                   const float* __restrict__ Wh,
                                   const float* __restrict__ Wd,
                                   __bf16* __restrict__ ws) {
  int idx = blockIdx.x * blockDim.x + threadIdx.x;
  if (idx >= WS_TOT) return;
  float v;
  if (idx < WS_WH)      v = Wi[idx];
  else if (idx < WS_WD) v = Wh[idx - WS_WH];
  else                  v = Wd[idx - WS_WD];
  ws[idx] = (__bf16)v;   // native v_cvt, RNE
}

__global__ __launch_bounds__(NTHREADS) void rnn_persistent_kernel(
    const float* __restrict__ x, const float* __restrict__ h0,
    const float* __restrict__ bh, const float* __restrict__ bd,
    const __bf16* __restrict__ ws, float* __restrict__ out) {
  // Double-buffered hidden state removes the mid-step WAR barrier.
  __shared__ __attribute__((aligned(16))) __bf16 hbuf[2][BB * H_STRIDE];
  __shared__ __attribute__((aligned(16))) __bf16 ainbuf[BB * IN_STRIDE];
  __shared__ __attribute__((aligned(16))) __bf16 obuf[BB * IN_STRIDE];

  const int tid  = threadIdx.x;
  const int lane = tid & 31;
  const int wave = tid >> 5;
  const int b0   = blockIdx.x * BB;

  const __bf16* wsWi = ws + WS_WI;
  const __bf16* wsWh = ws + WS_WH;
  const __bf16* wsWd = ws + WS_WD;

  // Phase-A tiling: z is 32x512 -> 2x32 tiles of 16x16; wave owns mt = wave&1,
  // N-tiles ng*8 .. ng*8+7 (ng = wave>>1). Phase-B: o is 32x128 -> 2x8 tiles;
  // wave owns mt and tiles nbB*2, nbB*2+1.
  const int mt  = wave & 1;
  const int ng  = wave >> 1;
  const int m0  = mt * 16;
  const int nbB = wave >> 1;
  const int l15 = lane & 15;
  const int lh8 = (lane >> 4) << 3;

  // Biases hoisted into registers for the whole time loop.
  float bhv[8];
#pragma unroll
  for (int j = 0; j < 8; ++j) bhv[j] = bh[(ng * 8 + j) * 16 + l15];
  float bdv[2];
#pragma unroll
  for (int j = 0; j < 2; ++j) bdv[j] = bd[(nbB * 2 + j) * 16 + l15];

  // Load initial hidden state h[-1] slice -> bf16 LDS (vectorized f32x4).
  for (int i = tid; i < BB * DLAT / 4; i += NTHREADS) {
    int r = i >> 7, c = (i & 127) << 2;
    f32x4 v = *(const f32x4*)(h0 + (size_t)(b0 + r) * DLAT + c);
    *(bf16x4*)(&hbuf[0][r * H_STRIDE + c]) = __builtin_convertvector(v, bf16x4);
  }
  __syncthreads();

  // Decode h (from hsrc) -> out[:, t, :] (f32) and obuf (bf16 feedback).
  auto decode_step = [&](const __bf16* hsrc, int t) {
    f32x8 acc[2];
#pragma unroll
    for (int j = 0; j < 2; ++j)
#pragma unroll
      for (int e = 0; e < 8; ++e) acc[j][e] = bdv[j];
#pragma unroll 4
    for (int kc = 0; kc < DLAT / 32; ++kc) {
      bf16x16 a = load_a_lds(hsrc, H_STRIDE, m0, kc * 32, lane);
#pragma unroll
      for (int j = 0; j < 2; ++j) {
        bf16x16 b = load_b_glb(wsWd, DLAT, (nbB * 2 + j) * 16, kc * 32, lane);
        acc[j] = wmma_bf16(a, b, acc[j]);
      }
    }
#pragma unroll
    for (int j = 0; j < 2; ++j) {
      const int c = (nbB * 2 + j) * 16 + l15;
#pragma unroll
      for (int e = 0; e < 8; ++e) {
        const int r = m0 + e + lh8;
        const float v = acc[j][e];
        out[((size_t)(b0 + r) * L_OUT + t) * DOUT + c] = v;
        obuf[r * IN_STRIDE + c] = (__bf16)v;
      }
    }
  };

  decode_step(hbuf[0], 0);  // out0 = decode(initial state)

  for (int s = 0; s < NSTEPS; ++s) {
    const __bf16* hcur = hbuf[s & 1];
    __bf16*       hnxt = hbuf[(s + 1) & 1];

    if (s < T_IN) {
      // Teacher forcing: stage x[s] slice as bf16 input (f32x4 vectorized).
      for (int i = tid; i < BB * DIN / 4; i += NTHREADS) {
        int r = i >> 5, c = (i & 31) << 2;
        f32x4 v = *(const f32x4*)(x + ((size_t)s * B_TOT + b0 + r) * DIN + c);
        *(bf16x4*)(&ainbuf[r * IN_STRIDE + c]) = __builtin_convertvector(v, bf16x4);
      }
    }
    __syncthreads();  // input/feedback staged; prior step fully retired

    const __bf16* inb = (s < T_IN) ? ainbuf : obuf;

    f32x8 acc[8];
#pragma unroll
    for (int j = 0; j < 8; ++j)
#pragma unroll
      for (int e = 0; e < 8; ++e) acc[j][e] = bhv[j];

    // z += a_in * Wi^T  (K = 128)
#pragma unroll
    for (int kc = 0; kc < DIN / 32; ++kc) {
      bf16x16 a = load_a_lds(inb, IN_STRIDE, m0, kc * 32, lane);
#pragma unroll
      for (int j = 0; j < 8; ++j) {
        bf16x16 b = load_b_glb(wsWi, DIN, (ng * 8 + j) * 16, kc * 32, lane);
        acc[j] = wmma_bf16(a, b, acc[j]);
      }
    }
    // z += h * Wh^T  (K = 512), Wh streamed from L2-resident bf16 copy
#pragma unroll 4
    for (int kc = 0; kc < DLAT / 32; ++kc) {
      bf16x16 a = load_a_lds(hcur, H_STRIDE, m0, kc * 32, lane);
#pragma unroll
      for (int j = 0; j < 8; ++j) {
        bf16x16 b = load_b_glb(wsWh, DLAT, (ng * 8 + j) * 16, kc * 32, lane);
        acc[j] = wmma_bf16(a, b, acc[j]);
      }
    }

    // h_next = relu(z) -> bf16 state (write other buffer: no WAR barrier needed)
#pragma unroll
    for (int j = 0; j < 8; ++j) {
      const int c = (ng * 8 + j) * 16 + l15;
#pragma unroll
      for (int e = 0; e < 8; ++e) {
        float v = fmaxf(acc[j][e], 0.f);
        hnxt[(m0 + e + lh8) * H_STRIDE + c] = (__bf16)v;
      }
    }
    __syncthreads();  // new h visible to all waves

    decode_step(hnxt, s + 1);
  }
}

extern "C" void kernel_launch(void* const* d_in, const int* in_sizes, int n_in,
                              void* d_out, int out_size, void* d_ws, size_t ws_size,
                              hipStream_t stream) {
  const float* x  = (const float*)d_in[0];
  const float* h0 = (const float*)d_in[1];
  const float* Wi = (const float*)d_in[2];
  const float* Wh = (const float*)d_in[3];
  const float* bh = (const float*)d_in[4];
  const float* Wd = (const float*)d_in[5];
  const float* bd = (const float*)d_in[6];
  // d_in[7] = sequence_out_length (512, fixed by the reference shapes)
  float* out = (float*)d_out;
  __bf16* ws = (__bf16*)d_ws;  // 786432 bytes of bf16 weights

  cvt_weights_kernel<<<(WS_TOT + 255) / 256, 256, 0, stream>>>(Wi, Wh, Wd, ws);
  rnn_persistent_kernel<<<NBLK, NTHREADS, 0, stream>>>(x, h0, bh, bd, ws, out);
}
